// TopKRouter_58798102282704
// MI455X (gfx1250) — matrix operand order repaired and checked by reading, not compiled, not run on hardware
//
#include <hip/hip_runtime.h>

#define N_TOKENS   32768
#define EMBED_DIM  1024
#define NUM_EXPERTS 64
#define ROWP (EMBED_DIM + 8)   // padded LDS row (ushorts): +16B kills bank conflicts

typedef __attribute__((ext_vector_type(16))) __bf16 v16bf;
typedef __attribute__((ext_vector_type(2)))  __bf16 v2bf;
typedef __attribute__((ext_vector_type(8)))  float  v8f;
typedef __attribute__((ext_vector_type(2)))  float  v2f;

struct __align__(16) U4 { unsigned int x, y, z, w; };
struct __align__(16) F4 { float x, y, z, w; };

union V16 {
  v16bf v;
  unsigned int u[8];
  U4 q[2];
};

// ---------------------------------------------------------------------------
// Kernel 1: split W_gate (fp32) into truncated-bf16 hi + RTE-bf16 lo residual
// ---------------------------------------------------------------------------
__global__ void wsplit_kernel(const float* __restrict__ W,
                              unsigned short* __restrict__ Whi,
                              unsigned short* __restrict__ Wlo) {
  int i = blockIdx.x * blockDim.x + threadIdx.x;
  float w = W[i];
  unsigned int u = __float_as_uint(w);
  Whi[i] = (unsigned short)(u >> 16);                       // truncation split
  float lo = w - __uint_as_float(u & 0xffff0000u);          // exact residual
  Wlo[i] = __builtin_bit_cast(unsigned short, (__bf16)lo);  // RTE
}

// ---------------------------------------------------------------------------
// Kernel 2: logits = x @ W^T via bf16x3 WMMA (hi*hi + hi*lo + lo*hi)
//   block = 256 thr (8 waves); wave -> 16 tokens x 64 experts
//   W_hi/W_lo staged in LDS with padded rows; A split to bf16 in registers
// ---------------------------------------------------------------------------
__global__ void __launch_bounds__(256) router_gemm_kernel(
    const float* __restrict__ x,
    const unsigned short* __restrict__ Whi,
    const unsigned short* __restrict__ Wlo,
    float* __restrict__ logits) {
  extern __shared__ unsigned short smem[];
  unsigned short* sWhi = smem;
  unsigned short* sWlo = smem + NUM_EXPERTS * ROWP;

  // Cooperative staging of W (row-major, padded rows) into LDS, b128 copies.
  {
    const U4* gh = (const U4*)Whi;
    const U4* gl = (const U4*)Wlo;
    for (int i = threadIdx.x; i < NUM_EXPERTS * (EMBED_DIM / 8); i += 256) {
      int row = i >> 7;          // EMBED_DIM/8 = 128 vec per row
      int col = i & 127;
      ((U4*)(sWhi + row * ROWP))[col] = gh[i];
      ((U4*)(sWlo + row * ROWP))[col] = gl[i];
    }
  }
  __syncthreads();

  const int lane = threadIdx.x & 31;
  const int wave = threadIdx.x >> 5;
  const int m    = lane & 15;    // token row (A) / expert col (B) within tile
  const int half = lane >> 4;    // K-half per ISA 16-bit operand layout
  const int tokBase = blockIdx.x * 128 + wave * 16;
  const float* xrow = x + (size_t)(tokBase + m) * EMBED_DIM;

  v8f acc[4] = {};               // 4 tiles of 16 experts, f32 accumulators

#pragma unroll 2
  for (int k0 = 0; k0 < EMBED_DIM; k0 += 32) {
    // A operand: lane(m,half) holds K = {k0+16h..+15} contiguous in VGPR0-7,
    // pairwise packed (matches B's contiguous K order).
    F4 a0 = *(const F4*)(xrow + k0 + 16 * half);
    F4 a1 = *(const F4*)(xrow + k0 + 16 * half + 4);
    F4 b0 = *(const F4*)(xrow + k0 + 16 * half + 8);
    F4 b1 = *(const F4*)(xrow + k0 + 16 * half + 12);
    float f[16] = {a0.x, a0.y, a0.z, a0.w, a1.x, a1.y, a1.z, a1.w,
                   b0.x, b0.y, b0.z, b0.w, b1.x, b1.y, b1.z, b1.w};
    V16 ahi, alo;
#pragma unroll
    for (int j = 0; j < 8; ++j) {
      float f0 = f[2 * j], f1 = f[2 * j + 1];
      unsigned int u0 = __float_as_uint(f0);
      unsigned int u1 = __float_as_uint(f1);
      // hi = truncated bf16 pair, packed in ONE v_perm_b32
      ahi.u[j] = __builtin_amdgcn_perm(u1, u0, 0x07060302u);
      // lo = exact residual, packed RTE convert (backend picks pk cvt)
      v2f lo;
      lo.x = f0 - __uint_as_float(u0 & 0xffff0000u);
      lo.y = f1 - __uint_as_float(u1 & 0xffff0000u);
      alo.u[j] = __builtin_bit_cast(unsigned int,
                                    __builtin_convertvector(lo, v2bf));
    }

#pragma unroll
    for (int t = 0; t < 4; ++t) {
      // B operand: lane n holds 16 contiguous bf16 of W row (t*16+n) at
      // k0+16*half -> direct 32-byte LDS read (two b128), bank-conflict-free.
      const int e = t * 16 + m;
      const U4* ph = (const U4*)(sWhi + e * ROWP + k0 + 16 * half);
      const U4* pl = (const U4*)(sWlo + e * ROWP + k0 + 16 * half);
      V16 bhi, blo;
      bhi.q[0] = ph[0]; bhi.q[1] = ph[1];
      blo.q[0] = pl[0]; blo.q[1] = pl[1];
      acc[t] = __builtin_amdgcn_wmma_f32_16x16x32_bf16(
          false, ahi.v, false, bhi.v, (short)0, acc[t], false, false);
      acc[t] = __builtin_amdgcn_wmma_f32_16x16x32_bf16(
          false, ahi.v, false, blo.v, (short)0, acc[t], false, false);
      acc[t] = __builtin_amdgcn_wmma_f32_16x16x32_bf16(
          false, alo.v, false, bhi.v, (short)0, acc[t], false, false);
    }
  }

  // C layout (ISA 7.12.2): VGPR r, lanes<16 -> M=r, lanes>=16 -> M=8+r; N=lane&15
  float* orow = logits + (size_t)(tokBase + half * 8) * NUM_EXPERTS;
#pragma unroll
  for (int t = 0; t < 4; ++t)
#pragma unroll
    for (int r = 0; r < 8; ++r)
      orow[(size_t)r * NUM_EXPERTS + t * 16 + m] = acc[t][r];
}

// ---------------------------------------------------------------------------
// Kernel 3: per-token softmax over 64 experts, top-2 (+renorm), aux-loss sums
// ---------------------------------------------------------------------------
__global__ void __launch_bounds__(256) router_topk_kernel(
    const float* __restrict__ logits,
    float* __restrict__ probs_out,   // (N,2)
    float* __restrict__ idx_out,     // (N,2) as float
    float* __restrict__ psum,        // [64] sum of probs over tokens
    float* __restrict__ fsum) {      // [64] top-2 pick counts
  __shared__ float sf[NUM_EXPERTS];
  __shared__ float sp[NUM_EXPERTS];
  if (threadIdx.x < NUM_EXPERTS) { sf[threadIdx.x] = 0.0f; sp[threadIdx.x] = 0.0f; }
  __syncthreads();

  const int t = blockIdx.x * blockDim.x + threadIdx.x;   // token id
  float l[NUM_EXPERTS];
  const F4* row = (const F4*)(logits + (size_t)t * NUM_EXPERTS);
#pragma unroll
  for (int i = 0; i < NUM_EXPERTS / 4; ++i) {
    F4 v = row[i];
    l[4 * i] = v.x; l[4 * i + 1] = v.y; l[4 * i + 2] = v.z; l[4 * i + 3] = v.w;
  }
  float mx = l[0];
#pragma unroll
  for (int e = 1; e < NUM_EXPERTS; ++e) mx = fmaxf(mx, l[e]);
  float s = 0.0f;
#pragma unroll
  for (int e = 0; e < NUM_EXPERTS; ++e) { l[e] = __expf(l[e] - mx); s += l[e]; }
  const float inv = 1.0f / s;

  float p1 = -1.0f, p2 = -1.0f; int i1 = 0, i2 = 0;
#pragma unroll
  for (int e = 0; e < NUM_EXPERTS; ++e) {
    float p = l[e] * inv;
    l[e] = p;
    if (p > p1)      { p2 = p1; i2 = i1; p1 = p; i1 = e; }  // strict > keeps
    else if (p > p2) { p2 = p;  i2 = e; }                   // lowest index on tie
  }

  // p-sum: wave-level butterfly reduce per expert, one LDS atomic per wave
#pragma unroll
  for (int e = 0; e < NUM_EXPERTS; ++e) {
    float v = l[e];
#pragma unroll
    for (int off = 16; off; off >>= 1) v += __shfl_xor(v, off, 32);
    if ((threadIdx.x & 31) == 0) atomicAdd(&sp[e], v);
  }
  atomicAdd(&sf[i1], 1.0f);
  atomicAdd(&sf[i2], 1.0f);

  const float den = p1 + p2;
  probs_out[2 * t]     = p1 / den;
  probs_out[2 * t + 1] = p2 / den;
  idx_out[2 * t]       = (float)i1;
  idx_out[2 * t + 1]   = (float)i2;

  __syncthreads();
  if (threadIdx.x < NUM_EXPERTS) {
    atomicAdd(&psum[threadIdx.x], sp[threadIdx.x]);
    atomicAdd(&fsum[threadIdx.x], sf[threadIdx.x]);
  }
}

// ---------------------------------------------------------------------------
// Kernel 4: aux_loss = E * sum_e (f_e/N) * (p_e/N)
// ---------------------------------------------------------------------------
__global__ void router_aux_kernel(const float* __restrict__ psum,
                                  const float* __restrict__ fsum,
                                  float* __restrict__ aux_out) {
  if (threadIdx.x == 0) {
    float a = 0.0f;
    for (int e = 0; e < NUM_EXPERTS; ++e) a += psum[e] * fsum[e];
    aux_out[0] = (float)NUM_EXPERTS * a / ((float)N_TOKENS * (float)N_TOKENS);
  }
}

// ---------------------------------------------------------------------------
extern "C" void kernel_launch(void* const* d_in, const int* in_sizes, int n_in,
                              void* d_out, int out_size, void* d_ws, size_t ws_size,
                              hipStream_t stream) {
  const float* x = (const float*)d_in[0];
  const float* W = (const float*)d_in[1];
  float* out = (float*)d_out;
  char* ws = (char*)d_ws;

  // workspace layout (~8.26 MB total)
  float* logits        = (float*)ws;                                   // 8 MB
  unsigned short* Whi  = (unsigned short*)(ws + (size_t)N_TOKENS * NUM_EXPERTS * 4);
  unsigned short* Wlo  = Whi + NUM_EXPERTS * EMBED_DIM;                // 128 KB each
  float* psum          = (float*)(Wlo + NUM_EXPERTS * EMBED_DIM);
  float* fsum          = psum + NUM_EXPERTS;

  hipMemsetAsync(psum, 0, 2 * NUM_EXPERTS * sizeof(float), stream);

  wsplit_kernel<<<NUM_EXPERTS * EMBED_DIM / 256, 256, 0, stream>>>(W, Whi, Wlo);

  const size_t ldsBytes = (size_t)2 * NUM_EXPERTS * ROWP * sizeof(unsigned short); // ~258 KB
  router_gemm_kernel<<<N_TOKENS / 128, 256, ldsBytes, stream>>>(x, Whi, Wlo, logits);

  router_topk_kernel<<<N_TOKENS / 256, 256, 0, stream>>>(
      logits, out, out + 2 * N_TOKENS, psum, fsum);

  router_aux_kernel<<<1, 64, 0, stream>>>(psum, fsum, out + 4 * N_TOKENS);
}
